// PNAEncoder_39565238731352
// MI455X (gfx1250) — matrix operand order repaired and checked by reading, not compiled, not run on hardware
//
#include <hip/hip_runtime.h>
#include <hip/hip_bf16.h>
#include <math.h>

#define N_NODES 30000
#define N_EDGES 480000
#define FDIM    128
#define LATENTD 64
#define NLAYERS 3
#define NGRAPHS 128

typedef __attribute__((ext_vector_type(16))) __bf16 v16bf;
typedef __attribute__((ext_vector_type(8)))  float  v8f;

#define CDIV(a,b) (((a)+(b)-1)/(b))

// ---------------------------------------------------------------------------
// WMMA helper (CDNA5: D = A(16x32 bf16) x B(32x16 bf16) + C(16x16 f32))
// ---------------------------------------------------------------------------
static __device__ inline v8f wmma_bf16(v16bf a, v16bf b, v8f c) {
  return __builtin_amdgcn_wmma_f32_16x16x32_bf16(false, a, false, b, (short)0, c,
                                                 false, false);
}

// A fragment 16x32 (MxK) bf16 from a row-major tile (stride must be mult of 8
// bf16 so both 16B runs stay aligned). Lane (m=l&15, hh=l>>4) holds
// K = [kk+8hh, +8) and [kk+16+8hh, +8): two contiguous 16B loads.
static __device__ inline v16bf load_a_frag(const __bf16* __restrict__ A,
                                           int stride, int row0, int kk,
                                           int lane) {
  int m = lane & 15, hh = lane >> 4;
  const uint4* q = (const uint4*)(A + (size_t)(row0 + m) * stride + kk + hh * 8);
  union { uint4 u[2]; v16bf v; } t;
  t.u[0] = q[0];
  t.u[1] = q[2];   // +16 bf16 = +32B = 2 uint4
  return t.v;
}

// A fragment gathered directly from a node-feature row in global memory.
static __device__ inline v16bf load_a_row(const __bf16* __restrict__ rowp) {
  const uint4* q = (const uint4*)rowp;
  union { uint4 u[2]; v16bf v; } t;
  t.u[0] = q[0];
  t.u[1] = q[2];
  return t.v;
}

// B fragment from fragment-major swizzled weights:
// Wswz[(kb*NT+nt)*32 + lane] holds that lane's 16 bf16 contiguously (32B).
static __device__ inline v16bf load_b_swz(const __bf16* __restrict__ Wswz,
                                          int kbnt, int lane) {
  const uint4* p = (const uint4*)(Wswz + ((size_t)kbnt * 32 + lane) * 16);
  union { uint4 u[2]; v16bf v; } t;
  t.u[0] = p[0];
  t.u[1] = p[1];
  return t.v;
}

// ---------------------------------------------------------------------------
// Small utility kernels
// ---------------------------------------------------------------------------
__global__ void k_fill_f32(float* p, float v, int n) {
  int i = blockIdx.x * 256 + threadIdx.x;
  if (i < n) p[i] = v;
}
__global__ void k_fill_u32(unsigned int* p, unsigned int v, int n) {
  int i = blockIdx.x * 256 + threadIdx.x;
  if (i < n) p[i] = v;
}
__global__ void k_deg(const int* __restrict__ ei, unsigned int* __restrict__ degu) {
  int i = blockIdx.x * 256 + threadIdx.x;
  if (i < N_EDGES) atomicAdd(&degu[ei[N_EDGES + i]], 1u);
}
__global__ void k_scalers(const unsigned int* __restrict__ degu,
                          const float* __restrict__ avgp,
                          float* __restrict__ invdeg, float* __restrict__ amp,
                          float* __restrict__ att) {
  int i = blockIdx.x * 256 + threadIdx.x;
  if (i >= N_NODES) return;
  float degc = fmaxf((float)degu[i], 1.0f);
  float ld = logf(degc + 1.0f);
  float avg = avgp[0];
  invdeg[i] = 1.0f / degc;
  amp[i] = ld / avg;
  att[i] = avg / ld;
}
__global__ void k_convert_x(const float* __restrict__ x, __bf16* __restrict__ xbf,
                            float* __restrict__ h) {
  int i = blockIdx.x * 256 + threadIdx.x;
  if (i >= N_NODES * FDIM) return;
  float v = x[i];
  xbf[i] = (__bf16)v;
  h[i] = v;
}
// W [rows][cols] f32 -> fragment-major swizzled bf16:
// out[(kb*NT+nt)*512 + lane*16 + v*2 + e] = W[nt*16 + (lane&15)]
//                                            [kb*32 + 2v + 8*(lane>>4) + 8*(v>>2) + e]
__global__ void k_w2bf_swz(const float* __restrict__ W, __bf16* __restrict__ out,
                           int rows, int cols, int NT) {
  int o = blockIdx.x * 256 + threadIdx.x;
  int total = (cols >> 5) * NT * 512;
  if (o >= total) return;
  int e = o & 1;
  int v = (o >> 1) & 7;
  int lane = (o >> 4) & 31;
  int rem = o >> 9;
  int nt = rem % NT, kb = rem / NT;
  int k = kb * 32 + 2 * v + 8 * (lane >> 4) + 8 * (v >> 2) + e;
  int n = nt * 16 + (lane & 15);
  out[o] = (__bf16)W[(size_t)n * cols + k];
}
__global__ void k_init_agg(float* __restrict__ s, float* __restrict__ s2,
                           float* __restrict__ mx, float* __restrict__ mn) {
  int i = blockIdx.x * 256 + threadIdx.x;
  if (i >= N_NODES * FDIM) return;
  s[i] = 0.0f; s2[i] = 0.0f;
  mx[i] = -__builtin_inff();
  mn[i] =  __builtin_inff();
}
// batch is sorted: accumulate per-graph runs locally, one atomic per run.
// 128 feature-threads per block, 64-node chunk per block.
__global__ __launch_bounds__(128) void k_pool(const float* __restrict__ h,
                                              const int* __restrict__ batch,
                                              float* __restrict__ pooled) {
  int f = threadIdx.x;
  int n0 = blockIdx.x * 64;
  int n1 = n0 + 64;
  if (n1 > N_NODES) n1 = N_NODES;
  int curg = batch[n0];
  float accv = 0.0f;
  for (int nd = n0; nd < n1; ++nd) {
    int g = batch[nd];
    if (g != curg) {
      atomicAdd(&pooled[curg * FDIM + f], accv);
      accv = 0.0f;
      curg = g;
    }
    accv += h[(size_t)nd * FDIM + f];
  }
  atomicAdd(&pooled[curg * FDIM + f], accv);
}

// ---------------------------------------------------------------------------
// Edge-message GEMM + fused PNA aggregation.
// h_e = Wpre @ [x_dst ; x_src] + bpre  -> atomics into sum/sumsq/max/min.
// 256 thr (8 waves), 128 edges/block, barrier-free main loop:
//   A: per-lane global b128 gathers from L2-resident x_bf rows
//   B: coalesced 32B/lane loads from fragment-major swizzled weights
// ---------------------------------------------------------------------------
__global__ __launch_bounds__(256) void k_edge_gemm(
    const __bf16* __restrict__ x_bf, const int* __restrict__ edge_index,
    const __bf16* __restrict__ Wswz, const float* __restrict__ bpre_l,
    float* __restrict__ sum, float* __restrict__ sumsq,
    float* __restrict__ amaxP, float* __restrict__ aminP) {
  __shared__ int dstS[128], srcS[128];

  int tid = threadIdx.x, lane = tid & 31, wave = tid >> 5;
  int ebase = blockIdx.x * 128;
  if (tid < 128) {
    srcS[tid] = edge_index[ebase + tid];
    dstS[tid] = edge_index[N_EDGES + ebase + tid];
  }
  __syncthreads();

  int m = lane & 15, hh = lane >> 4;
  int myDst = dstS[wave * 16 + m];
  int mySrc = srcS[wave * 16 + m];

  v8f acc[8] = {};
  for (int kb = 0; kb < 8; ++kb) {          // K = 256 = 8 x 32
    int node = (kb < 4) ? myDst : mySrc;    // [x_dst ; x_src] concat
    int koff = (kb & 3) * 32 + hh * 8;
    v16bf a = load_a_row(x_bf + (size_t)node * FDIM + koff);
#pragma unroll
    for (int nt = 0; nt < 8; ++nt) {
      v16bf b = load_b_swz(Wswz, kb * 8 + nt, lane);
      acc[nt] = wmma_bf16(a, b, acc[nt]);
    }
  }

  // Epilogue: bias + fused segment aggregation by dst.
  // Native CDNA5 float atomics: global_atomic_{add,max_num,min_num}_f32.
  int n = lane & 15;
#pragma unroll
  for (int nt = 0; nt < 8; ++nt) {
    int feat = nt * 16 + n;
    float bias = bpre_l[feat];
#pragma unroll
    for (int r = 0; r < 8; ++r) {
      int er = wave * 16 + hh * 8 + r;
      int d = dstS[er];
      float v = acc[nt][r] + bias;
      size_t ix = (size_t)d * FDIM + feat;
      atomicAdd(&sum[ix], v);
      atomicAdd(&sumsq[ix], v * v);
      unsafeAtomicMax(&amaxP[ix], v);
      unsafeAtomicMin(&aminP[ix], v);
    }
  }
}

// ---------------------------------------------------------------------------
// Fused post_nn (K=1664, A built on the fly) -> lin (K=128) -> ReLU.
// Wave-private LDS staging: thread tid builds exactly the rows its wave
// consumes, so the hot loop needs no __syncthreads.
// ---------------------------------------------------------------------------
__global__ __launch_bounds__(256) void k_post_gemm(
    const float* __restrict__ h, const float* __restrict__ sum,
    const float* __restrict__ sumsq, const float* __restrict__ amaxP,
    const float* __restrict__ aminP, const float* __restrict__ invdeg,
    const float* __restrict__ amp, const float* __restrict__ att,
    const __bf16* __restrict__ WpostSwz, const float* __restrict__ bpost_l,
    const __bf16* __restrict__ WlinSwz, const float* __restrict__ blin_l,
    float* __restrict__ h_out, __bf16* __restrict__ xbf_out) {
  const int AS = 80, TS = 136;              // strides: mult of 8 bf16 (16B runs)
  __shared__ __bf16 As[128 * 80];           // current 64-col A chunk
  __shared__ __bf16 Ts[128 * 136];          // GEMM1 output (bf16)
  __shared__ float ampS[128], attS[128], invS[128];

  int tid = threadIdx.x, lane = tid & 31, wave = tid >> 5;
  int base = blockIdx.x * 128;
  if (tid < 128) {
    int nd = base + tid;
    bool ok = nd < N_NODES;
    ampS[tid] = ok ? amp[nd] : 1.0f;
    attS[tid] = ok ? att[nd] : 1.0f;
    invS[tid] = ok ? invdeg[nd] : 1.0f;
  }
  __syncthreads();

  v8f acc[8] = {};
  int rowt = tid >> 1, half = tid & 1;      // wave-private: rowt in [16w,16w+16)
  int node = base + rowt;
  bool valid = node < N_NODES;
  float ampv = ampS[rowt], attv = attS[rowt], invd = invS[rowt];

  // ---- GEMM1: [x | (mean,min,max,std) x (1,amp,att)] @ Wpost^T, K=13F ----
  for (int k0 = 0; k0 < 13 * FDIM; k0 += 64) {
    {   // Build 32 A columns in registers, then 4x ds_store_b128
      union { __bf16 b[32]; uint4 u[4]; } tmp;
#pragma unroll
      for (int j = 0; j < 32; ++j) {
        int c = k0 + half * 32 + j;
        float v = 0.0f;
        if (valid) {
          if (c < FDIM) {
            v = h[(size_t)node * FDIM + c];
          } else {
            int jj = c - FDIM;
            int s = jj >> 9;            // scaler: 0=id 1=amp 2=att
            int a = (jj >> 7) & 3;      // agg: 0=mean 1=min 2=max 3=std
            int f = jj & 127;
            size_t ix = (size_t)node * FDIM + f;
            if (a == 0) {
              v = sum[ix] * invd;
            } else if (a == 1) {
              float t = aminP[ix]; v = (fabsf(t) < 1e30f) ? t : 0.0f;
            } else if (a == 2) {
              float t = amaxP[ix]; v = (fabsf(t) < 1e30f) ? t : 0.0f;
            } else {
              float mm = sum[ix] * invd;
              float q = sumsq[ix] * invd - mm * mm;
              v = sqrtf(fmaxf(q, 0.0f) + 1e-5f);
            }
            v *= (s == 0) ? 1.0f : ((s == 1) ? ampv : attv);
          }
        }
        tmp.b[j] = (__bf16)v;
      }
      uint4* dp = (uint4*)(As + rowt * AS + half * 32);
#pragma unroll
      for (int j = 0; j < 4; ++j) dp[j] = tmp.u[j];
    }
    int kb0 = k0 >> 5;
#pragma unroll
    for (int s = 0; s < 2; ++s) {
      v16bf a = load_a_frag(As, AS, wave * 16, s * 32, lane);
#pragma unroll
      for (int nt = 0; nt < 8; ++nt) {
        v16bf b = load_b_swz(WpostSwz, (kb0 + s) * 8 + nt, lane);
        acc[nt] = wmma_bf16(a, b, acc[nt]);
      }
    }
  }

  {   // GEMM1 epilogue -> Ts (wave-private rows)
    int n = lane & 15, hh = lane >> 4;
#pragma unroll
    for (int nt = 0; nt < 8; ++nt) {
      float bias = bpost_l[nt * 16 + n];
#pragma unroll
      for (int r = 0; r < 8; ++r)
        Ts[(wave * 16 + hh * 8 + r) * TS + nt * 16 + n] =
            (__bf16)(acc[nt][r] + bias);
    }
  }

  // ---- GEMM2: Ts @ Wlin^T, K = 128 ----
  const v8f vzero = {};
#pragma unroll
  for (int nt = 0; nt < 8; ++nt) acc[nt] = vzero;
#pragma unroll
  for (int kb = 0; kb < 4; ++kb) {
    v16bf a = load_a_frag(Ts, TS, wave * 16, kb * 32, lane);
#pragma unroll
    for (int nt = 0; nt < 8; ++nt) {
      v16bf b = load_b_swz(WlinSwz, kb * 8 + nt, lane);
      acc[nt] = wmma_bf16(a, b, acc[nt]);
    }
  }

  {   // bias + ReLU, write next-layer activations (f32 + bf16)
    int n = lane & 15, hh = lane >> 4;
#pragma unroll
    for (int nt = 0; nt < 8; ++nt) {
      int feat = nt * 16 + n;
      float bias = blin_l[feat];
#pragma unroll
      for (int r = 0; r < 8; ++r) {
        int nd = base + wave * 16 + hh * 8 + r;
        if (nd < N_NODES) {
          float v = fmaxf(acc[nt][r] + bias, 0.0f);
          h_out[(size_t)nd * FDIM + feat] = v;
          xbf_out[(size_t)nd * FDIM + feat] = (__bf16)v;
        }
      }
    }
  }
}

// ---------------------------------------------------------------------------
// BatchNorm over pooled [128 x 128] + FC head (WMMA, one block)
// ---------------------------------------------------------------------------
__global__ __launch_bounds__(256) void k_final(
    const float* __restrict__ pooled, const float* __restrict__ gamma,
    const float* __restrict__ beta, const __bf16* __restrict__ fcwSwz,
    const float* __restrict__ fcb, float* __restrict__ out) {
  __shared__ __bf16 Afs[128 * 136];
  __shared__ float sclS[128], sftS[128];
  int tid = threadIdx.x, lane = tid & 31, wave = tid >> 5;

  if (tid < FDIM) {
    float s = 0.0f, s2 = 0.0f;
    for (int g = 0; g < NGRAPHS; ++g) {
      float v = pooled[g * FDIM + tid];
      s += v; s2 += v * v;
    }
    float mu = s / NGRAPHS;
    float var = s2 / NGRAPHS - mu * mu;
    float scl = gamma[tid] * rsqrtf(var + 1e-5f);
    sclS[tid] = scl;
    sftS[tid] = beta[tid] - mu * scl;
  }
  __syncthreads();
  // Each wave normalizes the 16 rows it will consume (wave-private)
  for (int j = 0; j < 64; ++j) {
    int idx = lane + j * 32;                 // 16*128 elems per wave
    int g = wave * 16 + (idx >> 7), f = idx & 127;
    Afs[g * 136 + f] = (__bf16)(pooled[g * FDIM + f] * sclS[f] + sftS[f]);
  }

  int row0 = wave * 16;
  for (int nt = 0; nt < 4; ++nt) {
    v8f acc = {};
#pragma unroll
    for (int kb = 0; kb < 4; ++kb) {
      v16bf a = load_a_frag(Afs, 136, row0, kb * 32, lane);
      v16bf b = load_b_swz(fcwSwz, kb * 4 + nt, lane);
      acc = wmma_bf16(a, b, acc);
    }
    int n = lane & 15, hh = lane >> 4;
    float bias = fcb[nt * 16 + n];
#pragma unroll
    for (int r = 0; r < 8; ++r)
      out[(size_t)(row0 + hh * 8 + r) * LATENTD + nt * 16 + n] = acc[r] + bias;
  }
}

// ---------------------------------------------------------------------------
extern "C" void kernel_launch(void* const* d_in, const int* in_sizes, int n_in,
                              void* d_out, int out_size, void* d_ws,
                              size_t ws_size, hipStream_t stream) {
  const float* x     = (const float*)d_in[0];
  const int*   ei    = (const int*)d_in[1];
  const int*   batch = (const int*)d_in[2];
  const float* avgp  = (const float*)d_in[3];
  const float* Wpre  = (const float*)d_in[4];
  const float* bpre  = (const float*)d_in[5];
  const float* Wpost = (const float*)d_in[6];
  const float* bpost = (const float*)d_in[7];
  const float* Wlin  = (const float*)d_in[8];
  const float* blin  = (const float*)d_in[9];
  const float* gamma = (const float*)d_in[10];
  const float* beta  = (const float*)d_in[11];
  const float* fcw   = (const float*)d_in[12];
  const float* fcb   = (const float*)d_in[13];
  float* out = (float*)d_out;

  // Workspace carve-up
  char* ws = (char*)d_ws;
  size_t off = 0;
  auto carve = [&](size_t bytes) -> char* {
    char* p = ws + off;
    off = (off + bytes + 255) & ~(size_t)255;
    return p;
  };
  const size_t NF = (size_t)N_NODES * FDIM;
  float*        invdeg = (float*)carve(N_NODES * 4);
  float*        amp    = (float*)carve(N_NODES * 4);
  float*        att    = (float*)carve(N_NODES * 4);
  unsigned int* degu   = (unsigned int*)carve(N_NODES * 4);
  __bf16*       xbf    = (__bf16*)carve(NF * 2);
  float*        h      = (float*)carve(NF * 4);
  float*        sum    = (float*)carve(NF * 4);
  float*        sumsq  = (float*)carve(NF * 4);
  float*        amax   = (float*)carve(NF * 4);
  float*        amin   = (float*)carve(NF * 4);
  __bf16*       WpreB  = (__bf16*)carve((size_t)NLAYERS * 256 * 128 * 2);
  __bf16*       WpostB = (__bf16*)carve((size_t)NLAYERS * 1664 * 128 * 2);
  __bf16*       WlinB  = (__bf16*)carve((size_t)NLAYERS * 128 * 128 * 2);
  __bf16*       fcwB   = (__bf16*)carve((size_t)128 * 64 * 2);
  float*        pooled = (float*)carve((size_t)NGRAPHS * FDIM * 4);

  // Degree statistics + scalers
  k_fill_u32<<<CDIV(N_NODES, 256), 256, 0, stream>>>(degu, 0u, N_NODES);
  k_deg<<<CDIV(N_EDGES, 256), 256, 0, stream>>>(ei, degu);
  k_scalers<<<CDIV(N_NODES, 256), 256, 0, stream>>>(degu, avgp, invdeg, amp, att);
  k_convert_x<<<CDIV((int)NF, 256), 256, 0, stream>>>(x, xbf, h);

  // Weights -> fragment-major swizzled bf16
  for (int l = 0; l < NLAYERS; ++l) {
    k_w2bf_swz<<<CDIV(256 * 128, 256), 256, 0, stream>>>(
        Wpre + (size_t)l * 128 * 256, WpreB + (size_t)l * 256 * 128, 128, 256, 8);
    k_w2bf_swz<<<CDIV(1664 * 128, 256), 256, 0, stream>>>(
        Wpost + (size_t)l * 128 * 1664, WpostB + (size_t)l * 1664 * 128, 128,
        1664, 8);
    k_w2bf_swz<<<CDIV(128 * 128, 256), 256, 0, stream>>>(
        Wlin + (size_t)l * 128 * 128, WlinB + (size_t)l * 128 * 128, 128, 128, 8);
  }
  k_w2bf_swz<<<CDIV(128 * 64, 256), 256, 0, stream>>>(fcw, fcwB, 64, 128, 4);

  // PNA layers
  for (int l = 0; l < NLAYERS; ++l) {
    k_init_agg<<<CDIV((int)NF, 256), 256, 0, stream>>>(sum, sumsq, amax, amin);
    k_edge_gemm<<<N_EDGES / 128, 256, 0, stream>>>(
        xbf, ei, WpreB + (size_t)l * 256 * 128, bpre + l * 128, sum, sumsq, amax,
        amin);
    k_post_gemm<<<CDIV(N_NODES, 128), 256, 0, stream>>>(
        h, sum, sumsq, amax, amin, invdeg, amp, att,
        WpostB + (size_t)l * 1664 * 128, bpost + l * 128,
        WlinB + (size_t)l * 128 * 128, blin + l * 128, h, xbf);
  }

  // Pool + BN + FC head
  k_fill_f32<<<CDIV(NGRAPHS * FDIM, 256), 256, 0, stream>>>(
      pooled, 0.0f, NGRAPHS * FDIM);
  k_pool<<<CDIV(N_NODES, 64), 128, 0, stream>>>(h, batch, pooled);
  k_final<<<1, 256, 0, stream>>>(pooled, gamma, beta, fcwB, fcb, out);
}
